// ChannelwiseSpatialMHSA_19061064860255
// MI455X (gfx1250) — matrix operand order repaired and checked by reading, not compiled, not run on hardware
//
#include <hip/hip_runtime.h>
#include <hip/hip_bf16.h>

typedef __attribute__((ext_vector_type(16))) _Float16 v16h;
typedef __attribute__((ext_vector_type(8)))  _Float16 v8h;
typedef __attribute__((ext_vector_type(2)))  _Float16 v2h;
typedef __attribute__((ext_vector_type(8)))  float    v8f;

#define B_  2
#define HH_ 32
#define WW_ 32
#define C_  32
#define DM_ 64
#define NH_ 4
#define DH_ 16
#define S_  1024   // HH_*WW_
#define NSEQ_ 64   // B_*C_

// ---------------------------------------------------------------------------
// Kernel A: per-(sequence, head) one-pass flash attention.
//
// embed_w is [64,1] => E is rank-1 => Q/K/V rows are exact scalar multiples
// of per-head 16-vectors: Q[s,d]=t[s]*qv[d], etc.  Consequences used here:
//  * Scores are rank-1: S[s,t'] = coef[s]*t[t'], coef[s]=a_h*t[s] with
//    a_h = (qv.kv)/4 (log2e folded in).  Each lane's score is one v_fmac
//    in full f32 -- cheaper AND more accurate than a padded f16 WMMA.
//  * A per-row softmax bound is analytic: m_row = max(coef*gmax, coef*gmin)
//    + 1.0 (exact power-of-two margin, cancels in o/l).  One-pass softmax:
//    no cross-lane reductions, no online rescaling.
//  * P = exp(S-m) is NOT rank-1 (exp is nonlinear) -- the P.V contraction is
//    the one genuine GEMM, kept on v_wmma_f32_16x16x32_f16 at full K=32,
//    plus a P.ones WMMA that yields row sums with no reduction anywhere.
//
// Inner iteration (32 K-columns):
//   2x ds_load_b32 (t columns), 16x v_fmac (scores, bound fused),
//   16x raw v_exp_f32, 8x v_cvt_pk + 8x ds_store_b32 (adjacent-column P
//   bounce), 4x ds_load_b128 (P A-fragment + V B-fragment),
//   1x WMMA P.V + 1x WMMA P.1.
// ---------------------------------------------------------------------------
__global__ __launch_bounds__(128)
void attn_wmma_kernel(const float* __restrict__ x,
                      const float* __restrict__ embed_w,
                      const float* __restrict__ q_w,
                      const float* __restrict__ k_w,
                      const float* __restrict__ v_w,
                      float* __restrict__ Yh)
{
    const int n    = blockIdx.x >> 2;     // sequence 0..63  (n = b*32 + c)
    const int head = blockIdx.x & 3;      // head 0..3
    const int b    = n >> 5;
    const int c    = n & 31;

    const int tid   = threadIdx.x;
    const int lane  = tid & 31;
    const int wave  = tid >> 5;           // 0..3
    const int lo    = lane & 15;          // lane within 16-lane half
    const int hi    = lane >> 4;          // 0: lanes 0-15, 1: lanes 16-31
    const int kbase = hi * 8;             // unused; A and B fragments both use koff
    const int koff  = hi * 16;            // B-fragment K offset for this half

    __shared__ float    tv[S_];                        // t[n,:]        4KB
    __shared__ __align__(32) _Float16 Vt16[DH_][S_];   // V^T f16 table 32KB
    __shared__ __align__(32) _Float16 Pbuf[4][16][32]; // P bounce       4KB
    __shared__ float    qvs[DH_], kvs[DH_], vvs[DH_];
    __shared__ float    pmax[128], pmin[128];
    __shared__ float    gred[3];                       // gmax, gmin, a_h

    // t[n, s=hh*32+ww] = x[b, hh, ww, c]; track per-thread max/min on the fly
    float lmax = -3.0e38f, lmin = 3.0e38f;
    for (int t = tid; t < S_; t += 128) {
        int hh = t >> 5, ww = t & 31;
        float v = x[((b * HH_ + hh) * WW_ + ww) * C_ + c];
        tv[t] = v;
        lmax = fmaxf(lmax, v);
        lmin = fminf(lmin, v);
    }
    pmax[tid] = lmax;
    pmin[tid] = lmin;

    // Head projection vectors; 1/sqrt(16)*log2(e) folded into qv so scores
    // are in base-2 units consumed directly by v_exp_f32.
    if (tid < 48) {
        int which = tid >> 4, dd = tid & 15;
        const float* wrow = (which == 0 ? q_w : which == 1 ? k_w : v_w)
                          + (head * DH_ + dd) * DM_;
        float acc = 0.f;
        #pragma unroll 8
        for (int k = 0; k < DM_; ++k) acc += wrow[k] * embed_w[k];
        if      (which == 0) qvs[dd] = acc * 0.25f * 1.44269504088896340736f;
        else if (which == 1) kvs[dd] = acc;
        else                 vvs[dd] = acc;
    }
    __syncthreads();

    // Thread 0 reduces gmax/gmin and a_h while the block builds the f16
    // V^T fragment table (vectorized 32B stores).
    if (tid == 0) {
        float gm = pmax[0], gn = pmin[0];
        for (int i = 1; i < 128; ++i) {
            gm = fmaxf(gm, pmax[i]);
            gn = fminf(gn, pmin[i]);
        }
        float ahv = 0.f;
        #pragma unroll
        for (int d = 0; d < DH_; ++d) ahv += qvs[d] * kvs[d];
        gred[0] = gm; gred[1] = gn; gred[2] = ahv;
    }
    for (int i = tid; i < DH_ * (S_ / 16); i += 128) { // Vt16[d][tb..tb+15]
        const int d = i >> 6, tb = (i & 63) * 16;
        const float vd = vvs[d];
        v16h chunk;
        #pragma unroll
        for (int j = 0; j < 16; ++j) chunk[j] = (_Float16)(tv[tb + j] * vd);
        *(v16h*)&Vt16[d][tb] = chunk;
    }
    __syncthreads();

    const float gmax = gred[0], gmin = gred[1], ah = gred[2];

    v16h bones;
    #pragma unroll
    for (int j = 0; j < 16; ++j) bones[j] = (_Float16)1.0f;

    // Each wave owns Q tiles wave, wave+4, ... (16 tiles of 16 rows).
    for (int qt = wave; qt < S_ / 16; qt += 4) {
        const int s0 = qt * 16;

        // Per-row score coefficient and fused analytic bound (base-2).
        float coef[8], cm[8];
        #pragma unroll
        for (int r = 0; r < 8; ++r) {
            coef[r] = ah * tv[s0 + r + 8 * hi];
            cm[r]   = -(fmaxf(coef[r] * gmax, coef[r] * gmin) + 1.0f);
        }

        v8f o  = {};
        v8f ls = {};

        for (int kt = 0; kt < S_ / 32; ++kt) {
            const int t0 = kt * 32;

            // Rank-1 scores in C-fragment layout; each lane owns two
            // ADJACENT columns so the f16 pair packs into one b32 store.
            const float tc0 = tv[t0 + 2 * lo];
            const float tc1 = tv[t0 + 2 * lo + 1];

            // One-pass probabilities: p = 2^(coef*t[col] - m_row).
            #pragma unroll
            for (int r = 0; r < 8; ++r) {
                const float p0 = __builtin_amdgcn_exp2f(fmaf(coef[r], tc0, cm[r]));
                const float p1 = __builtin_amdgcn_exp2f(fmaf(coef[r], tc1, cm[r]));
                v2h pp;
                pp[0] = (_Float16)p0;
                pp[1] = (_Float16)p1;
                *(v2h*)&Pbuf[wave][r + 8 * hi][2 * lo] = pp;
            }
            asm volatile("" ::: "memory");   // DS ops are in-order per wave

            // P in A-fragment layout (full 16x32): two aligned 16B reads.
            const v8h pl0 = *(const v8h*)&Pbuf[wave][lo][koff];
            const v8h pl1 = *(const v8h*)&Pbuf[wave][lo][koff + 8];
            const v16h ap = __builtin_shufflevector(pl0, pl1,
                                0, 1, 2, 3, 4, 5, 6, 7,
                                8, 9, 10, 11, 12, 13, 14, 15);

            // V B-fragment: contiguous 32B read from the V^T table.
            const v16h bv = *(const v16h*)&Vt16[lo][t0 + koff];

            // The genuine GEMM: O += P.V (full K=32, f32 accumulate).
            o  = __builtin_amdgcn_wmma_f32_16x16x32_f16(
                     false, ap, false, bv, (short)0, o, false, false);
            // Row sums: every output column of P x ones holds rowsum(P).
            ls = __builtin_amdgcn_wmma_f32_16x16x32_f16(
                     false, ap, false, bones, (short)0, ls, false, false);
        }

        // Normalize and emit Yh[n][head][s0+M][d=lo].  ls[r] holds the full
        // row sum in every lane; v_rcp_f32 (~1 ulp) instead of IEEE divide
        // (error far below the f16 P quantization already present).
        #pragma unroll
        for (int r = 0; r < 8; ++r) {
            const int srow = s0 + r + 8 * hi;
            Yh[((n * NH_ + head) * S_ + srow) * DH_ + lo] =
                o[r] * __builtin_amdgcn_rcpf(ls[r]);
        }
    }
}

// ---------------------------------------------------------------------------
// Kernel B: output projection + channel merge (deterministic, no atomics).
// out[b,s,o] = sum_c merge[c] * sum_{h,d} Yh[b*32+c][h][s][d] * o_w[o, h*16+d]
// ---------------------------------------------------------------------------
__global__ __launch_bounds__(64)
void merge_kernel(const float* __restrict__ Yh,
                  const float* __restrict__ o_w,
                  const float* __restrict__ merge_w,
                  float* __restrict__ out)
{
    const int bs = blockIdx.x;           // b*1024 + s
    const int b  = bs >> 10;
    const int s  = bs & 1023;
    const int tid = threadIdx.x;         // output channel o

    __shared__ float Ys[C_ * DM_];       // [c][dcol=h*16+d]  8KB
    __shared__ float ows[DM_ * DM_];     // o_w               16KB
    __shared__ float mw[C_];

    for (int i = tid; i < C_ * DM_; i += 64) {
        const int cc = i >> 6, dcol = i & 63;
        const int h = dcol >> 4, d = dcol & 15;
        Ys[i] = Yh[(((b * C_ + cc) * NH_ + h) * S_ + s) * DH_ + d];
    }
    for (int i = tid; i < DM_ * DM_; i += 64) ows[i] = o_w[i];
    if (tid < C_) mw[tid] = merge_w[tid];
    __syncthreads();

    float acc = 0.f;
    for (int cc = 0; cc < C_; ++cc) {
        float a2 = 0.f;
        #pragma unroll 8
        for (int dcol = 0; dcol < DM_; ++dcol)
            a2 += Ys[cc * DM_ + dcol] * ows[tid * DM_ + dcol];
        acc += mw[cc] * a2;
    }
    out[bs * 64 + tid] = acc;
}

extern "C" void kernel_launch(void* const* d_in, const int* in_sizes, int n_in,
                              void* d_out, int out_size, void* d_ws, size_t ws_size,
                              hipStream_t stream) {
    const float* x       = (const float*)d_in[0];
    const float* embed_w = (const float*)d_in[1];
    const float* q_w     = (const float*)d_in[2];
    const float* k_w     = (const float*)d_in[3];
    const float* v_w     = (const float*)d_in[4];
    const float* o_w     = (const float*)d_in[5];
    const float* merge_w = (const float*)d_in[6];
    float* Yh  = (float*)d_ws;           // 64*4*1024*16 f32 = 16 MB
    float* out = (float*)d_out;

    attn_wmma_kernel<<<NSEQ_ * NH_, 128, 0, stream>>>(x, embed_w, q_w, k_w, v_w, Yh);
    merge_kernel<<<B_ * S_, 64, 0, stream>>>(Yh, o_w, merge_w, out);
}
